// LESSR_8306466751033
// MI455X (gfx1250) — matrix operand order repaired and checked by reading, not compiled, not run on hardware
//
#include <hip/hip_runtime.h>
#include <math.h>

// ---------------------------------------------------------------------------
// LESSR forward for MI455X (gfx1250, wave32, WMMA).
// Heavy matmuls -> v_wmma_f32_16x16x32_bf16, double-buffered LDS fed by
// CDNA5 async global->LDS DMA (global_load_async_to_lds_b128 / ASYNCcnt).
// ---------------------------------------------------------------------------

typedef unsigned short u16;
typedef unsigned int   u32;
typedef __attribute__((ext_vector_type(16))) __bf16 v16bf;
typedef __attribute__((ext_vector_type(8)))  float  v8f;

static constexpr int Bg  = 2048;
static constexpr int Lg  = 20;
static constexpr int Dd  = 128;
static constexpr int DEGc= 4;
static constexpr int Vv  = 50000;
static constexpr int Ng  = Bg * Lg;      // 40960 nodes
static constexpr int FW  = 512;          // final feat width D*(NL+1)
static constexpr float EPSc = 1e-5f;
static constexpr int LP  = 40;           // LDS row pitch (elems): 80B, conflict-free

#define DEV __device__ __forceinline__

DEV u16 f2bf(float f) {                      // round-to-nearest-even f32 -> bf16
    u32 u = __float_as_uint(f);
    u32 r = u + 0x7FFFu + ((u >> 16) & 1u);
    return (u16)(r >> 16);
}
DEV float sigm(float x) { return 1.0f / (1.0f + __expf(-x)); }

// ============================= WMMA GEMM ====================================
// C[M,Nd] (f32) = A[M,K] (bf16 row-major, lda) @ W[Nd,K]^T (bf16 row-major, ldw)
// Block tile 128x128, K-step 32, 8 waves, each wave 32x64 (8 WMMAs/chunk).
// A/W tiles staged by async DMA into double-buffered LDS.
union FragAB { uint4 u[2]; v16bf v; };

// Stage one 128x32-bf16 tile (512 x 16B chunks, 2 per thread) via async DMA.
// Rows are clamped (never branch -> EXEC stays all-ones); clamped duplicates
// only feed output columns >= Nd, which are never stored.
DEV void async_stage(const u16* __restrict__ g, int ld, int row0, int rowClamp,
                     int k0, u16* lds) {
    const int t = threadIdx.x;
    #pragma unroll
    for (int it = 0; it < 2; ++it) {
        int c  = t + (it << 8);          // 0..511
        int r  = c >> 2;                 // tile row 0..127
        int co = (c & 3) << 3;           // elem col 0,8,16,24
        int gr = row0 + r; gr = gr < rowClamp ? gr : rowClamp;
        const u16* gp = g + (size_t)gr * ld + k0 + co;
        unsigned la = (unsigned)(size_t)(lds + r * LP + co);
        asm volatile("global_load_async_to_lds_b128 %0, %1, off"
                     :: "v"(la), "v"(gp) : "memory");
    }
}

__global__ __launch_bounds__(256)
void gemm_bf16_wmma(const u16* __restrict__ A, int lda,
                    const u16* __restrict__ W, int ldw,
                    float* __restrict__ C, int ldc,
                    int M, int Nd, int K) {
    __shared__ __align__(16) u16 As[2][128 * LP];
    __shared__ __align__(16) u16 Bs[2][128 * LP];

    const int t    = threadIdx.x;
    const int w    = t >> 5;
    const int lane = t & 31;
    const int ml   = lane & 15;         // A row / B col within fragment
    const int hh   = lane >> 4;         // half-wave select
    const int wm   = (w >> 1) << 5;     // wave M offset: 0/32/64/96
    const int wn   = (w & 1) << 6;      // wave N offset: 0/64

    const int m0 = blockIdx.y * 128;    // M is a multiple of 128 for all calls
    const int n0 = blockIdx.x * 128;    // Nd may be ragged (guarded)

    v8f acc[2][4] = {};
    const int KC = K >> 5;

    async_stage(A, lda, m0, M - 1, 0, &As[0][0]);
    async_stage(W, ldw, n0, Nd - 1, 0, &Bs[0][0]);

    for (int kc = 0; kc < KC; ++kc) {
        const int buf = kc & 1;
        if (kc + 1 < KC) {
            async_stage(A, lda, m0, M - 1, (kc + 1) << 5, &As[buf ^ 1][0]);
            async_stage(W, ldw, n0, Nd - 1, (kc + 1) << 5, &Bs[buf ^ 1][0]);
            asm volatile("s_wait_asynccnt 0x4" ::: "memory");  // stage kc done
        } else {
            asm volatile("s_wait_asynccnt 0x0" ::: "memory");
        }
        __syncthreads();                 // all waves' DMA for buf complete

        const u16* a_ = &As[buf][0];
        const u16* b_ = &Bs[buf][0];
        // Preload ALL fragments (12 x ds_load_b128), then fire 8 WMMAs
        // back-to-back so the matrix pipe is not serialized on DScnt.
        FragAB a0, a1, bfr[4];
        a0.u[0] = *(const uint4*)&a_[(wm + ml) * LP + 8 * hh];
        a0.u[1] = *(const uint4*)&a_[(wm + ml) * LP + 16 + 8 * hh];
        a1.u[0] = *(const uint4*)&a_[(wm + 16 + ml) * LP + 8 * hh];
        a1.u[1] = *(const uint4*)&a_[(wm + 16 + ml) * LP + 16 + 8 * hh];
        #pragma unroll
        for (int j = 0; j < 4; ++j) {
            bfr[j].u[0] = *(const uint4*)&b_[(wn + 16 * j + ml) * LP + 8 * hh];
            bfr[j].u[1] = *(const uint4*)&b_[(wn + 16 * j + ml) * LP + 16 + 8 * hh];
        }
        #pragma unroll
        for (int j = 0; j < 4; ++j) {
            acc[0][j] = __builtin_amdgcn_wmma_f32_16x16x32_bf16(
                            false, a0.v, false, bfr[j].v, (short)0, acc[0][j], false, false);
            acc[1][j] = __builtin_amdgcn_wmma_f32_16x16x32_bf16(
                            false, a1.v, false, bfr[j].v, (short)0, acc[1][j], false, false);
        }
        __syncthreads();                 // buf free for reuse at kc+2
    }

    // C/D layout: VGPR r -> M = r + 8*hh ; lane ml -> N
    #pragma unroll
    for (int i = 0; i < 2; ++i) {
        const int rowBase = m0 + wm + 16 * i + 8 * hh;
        #pragma unroll
        for (int j = 0; j < 4; ++j) {
            const int col = n0 + wn + 16 * j + ml;
            if (col < Nd) {
                #pragma unroll
                for (int r = 0; r < 8; ++r)
                    C[(size_t)(rowBase + r) * ldc + col] = acc[i][j][r];
            }
        }
    }
}

// ============================ elementwise kernels ===========================

__global__ void k_cvt_bf16(const float* __restrict__ s, u16* __restrict__ d, int n) {
    int i = blockIdx.x * blockDim.x + threadIdx.x;
    if (i < n) d[i] = f2bf(s[i]);
}
__global__ void k_fill_f32(float* p, float v, int n) {
    int i = blockIdx.x * blockDim.x + threadIdx.x;
    if (i < n) p[i] = v;
}
__global__ void k_fill_u16(u16* p, u16 v, int n) {
    int i = blockIdx.x * blockDim.x + threadIdx.x;
    if (i < n) p[i] = v;
}

// embedding max_norm=1 renorm; emits f32 + bf16 copies. 1 block = 1 row.
__global__ void k_renorm_emb(const float* __restrict__ emb,
                             float* __restrict__ of, u16* __restrict__ ob) {
    int r = blockIdx.x, h = threadIdx.x;           // 128 threads
    __shared__ float red[128];
    float v = emb[(size_t)r * Dd + h];
    red[h] = v * v; __syncthreads();
    for (int s = 64; s > 0; s >>= 1) { if (h < s) red[h] += red[h + s]; __syncthreads(); }
    float nrm = sqrtf(red[0]);
    float sc = (nrm > 1.0f) ? 1.0f / fmaxf(nrm, 1e-12f) : 1.0f;
    float o = v * sc;
    of[(size_t)r * Dd + h] = o;
    ob[(size_t)r * Dd + h] = f2bf(o);
}

__global__ void k_gather_emb(const float* __restrict__ embn,
                             const int* __restrict__ iid,
                             float* __restrict__ feat, int total) {
    int i = blockIdx.x * blockDim.x + threadIdx.x;
    if (i >= total) return;
    int n = i >> 7, c = i & 127;
    feat[(size_t)n * FW + (FW - Dd) + c] = embn[(size_t)iid[n] * Dd + c];
}

// column-wise training-mode BN stats: scale = g*rsqrt(var+eps), shift = b-mean*scale
__global__ void k_bn_stats(const float* __restrict__ x, int stride, int nrows,
                           const float* __restrict__ g, const float* __restrict__ b,
                           float* __restrict__ scale, float* __restrict__ shift) {
    int c = blockIdx.x, t = threadIdx.x;           // 256 threads per column
    __shared__ float s1[256], s2[256];
    float a = 0.f, q = 0.f;
    for (int r = t; r < nrows; r += 256) {
        float v = x[(size_t)r * stride + c];
        a += v; q += v * v;
    }
    s1[t] = a; s2[t] = q; __syncthreads();
    for (int s = 128; s > 0; s >>= 1) {
        if (t < s) { s1[t] += s1[t + s]; s2[t] += s2[t + s]; }
        __syncthreads();
    }
    if (t == 0) {
        float mean = s1[0] / nrows;
        float var  = s2[0] / nrows - mean * mean;   // biased, as torch BN training
        float sc = g[c] * rsqrtf(var + EPSc);
        scale[c] = sc;
        shift[c] = b[c] - mean * sc;
    }
}

__global__ void k_bn_apply(const float* __restrict__ x, int in_stride, int ncols,
                           const float* __restrict__ scale, const float* __restrict__ shift,
                           u16* __restrict__ yb, float* __restrict__ yf, int total) {
    int i = blockIdx.x * blockDim.x + threadIdx.x;
    if (i >= total) return;
    int r = i / ncols, c = i - r * ncols;
    float v = x[(size_t)r * in_stride + c] * scale[c] + shift[c];
    yb[i] = f2bf(v);
    if (yf) yf[i] = v;
}

// gather mailbox rows: dst[n,:] = src[mg_src[n*DEG+t], :]
__global__ void k_gather_rows_bf(const u16* __restrict__ src, const int* __restrict__ idx,
                                 int t, int width, u16* __restrict__ dst, int total) {
    int i = blockIdx.x * blockDim.x + threadIdx.x;
    if (i >= total) return;
    int n = i / width, c = i - n * width;
    dst[i] = src[(size_t)idx[n * DEGc + t] * width + c];
}

__global__ void k_gather_last_bf(const u16* __restrict__ src, u16* __restrict__ dst,
                                 int width, int total) {
    int i = blockIdx.x * blockDim.x + threadIdx.x;
    if (i >= total) return;
    int b = i / width, c = i - b * width;
    dst[i] = src[((size_t)b * Lg + (Lg - 1)) * width + c];
}

// fused GRU gate update: h = (1-z)*n + z*h  (writes f32 + bf16 state)
__global__ void k_gru_update(const float* __restrict__ gi, const float* __restrict__ gh,
                             const float* __restrict__ bih, const float* __restrict__ bhh,
                             float* __restrict__ h, u16* __restrict__ hb,
                             int d, int total) {
    int i = blockIdx.x * blockDim.x + threadIdx.x;
    if (i >= total) return;
    int n = i / d, c = i - n * d;
    size_t base = (size_t)n * 3 * d;
    float ir = gi[base + c]         + bih[c];
    float iz = gi[base + d + c]     + bih[d + c];
    float in = gi[base + 2 * d + c] + bih[2 * d + c];
    float hr = gh[base + c]         + bhh[c];
    float hz = gh[base + d + c]     + bhh[d + c];
    float hn = gh[base + 2 * d + c] + bhh[2 * d + c];
    float r  = sigm(ir + hr);
    float z  = sigm(iz + hz);
    float nn = tanhf(in + r * hn);
    float hv = (1.0f - z) * nn + z * h[i];
    h[i]  = hv;
    hb[i] = f2bf(hv);
}

// out = prelu(a (+ b)) written into feat columns [colofs, colofs+128)
__global__ void k_prelu_store(const float* __restrict__ a, const float* __restrict__ b,
                              const float* __restrict__ al, float* __restrict__ feat,
                              int colofs, int total) {
    int i = blockIdx.x * blockDim.x + threadIdx.x;
    if (i >= total) return;
    int n = i >> 7, c = i & 127;
    float v = a[i];
    if (b) v += b[i];
    v = (v > 0.f) ? v : al[c] * v;
    feat[(size_t)n * FW + colofs + c] = v;
}

// SGAT: per-dst edge attention + softmax over DEG.  1 block = 1 node, 128 thr.
__global__ void k_sgat(const float* __restrict__ q, const float* __restrict__ k,
                       const float* __restrict__ v, const float* __restrict__ bq,
                       const float* __restrict__ we, const int* __restrict__ sg,
                       float* __restrict__ out) {
    int n = blockIdx.x, h = threadIdx.x;
    __shared__ float red[128];
    __shared__ float logit[DEGc];
    __shared__ float alpha[DEGc];
    __shared__ int   src[DEGc];
    if (h < DEGc) src[h] = sg[n * DEGc + h];
    __syncthreads();
    float kn = k[(size_t)n * Dd + h];
    float qb = bq[h], weh = we[h];
    for (int t = 0; t < DEGc; ++t) {
        float e = sigm(q[(size_t)src[t] * Dd + h] + qb + kn);
        red[h] = e * weh; __syncthreads();
        for (int s = 64; s > 0; s >>= 1) { if (h < s) red[h] += red[h + s]; __syncthreads(); }
        if (h == 0) logit[t] = red[0];
        __syncthreads();
    }
    if (h == 0) {
        float mx = fmaxf(fmaxf(logit[0], logit[1]), fmaxf(logit[2], logit[3]));
        float s = 0.f, e[DEGc];
        for (int t = 0; t < DEGc; ++t) { e[t] = __expf(logit[t] - mx); s += e[t]; }
        for (int t = 0; t < DEGc; ++t) alpha[t] = e[t] / s;
    }
    __syncthreads();
    float o = 0.f;
    for (int t = 0; t < DEGc; ++t) o += alpha[t] * v[(size_t)src[t] * Dd + h];
    out[(size_t)n * Dd + h] = o;
}

// readout edge score: e[n] = dot(sigmoid(fu[n]+fv[b]+bv), we)
__global__ void k_ro_score(const float* __restrict__ fu, const float* __restrict__ fv,
                           const float* __restrict__ bv, const float* __restrict__ we,
                           float* __restrict__ esc) {
    int n = blockIdx.x, h = threadIdx.x;
    int b = n / Lg;
    __shared__ float red[128];
    float v = sigm(fu[(size_t)n * Dd + h] + fv[(size_t)b * Dd + h] + bv[h]) * we[h];
    red[h] = v; __syncthreads();
    for (int s = 64; s > 0; s >>= 1) { if (h < s) red[h] += red[h + s]; __syncthreads(); }
    if (h == 0) esc[n] = red[0];
}

// per-graph node softmax + weighted sum of xr (512 dims). 1 block = 1 graph.
__global__ void k_ro_attnsum(const float* __restrict__ esc, const float* __restrict__ xr,
                             float* __restrict__ srg, u16* __restrict__ srgb) {
    int b = blockIdx.x, t = threadIdx.x;          // 256 threads
    __shared__ float a[Lg];
    if (t == 0) {
        float mx = -1e30f;
        for (int l = 0; l < Lg; ++l) mx = fmaxf(mx, esc[b * Lg + l]);
        float s = 0.f;
        for (int l = 0; l < Lg; ++l) { a[l] = __expf(esc[b * Lg + l] - mx); s += a[l]; }
        float inv = 1.0f / s;
        for (int l = 0; l < Lg; ++l) a[l] *= inv;
    }
    __syncthreads();
    for (int c = t; c < FW; c += 256) {
        float acc = 0.f;
        for (int l = 0; l < Lg; ++l) acc += a[l] * xr[((size_t)b * Lg + l) * FW + c];
        srg[(size_t)b * FW + c]  = acc;
        srgb[(size_t)b * FW + c] = f2bf(acc);
    }
}

// sr = concat(feat[last] (512), sr_g_out (128)) -> [B, 640]
__global__ void k_concat_sr(const float* __restrict__ feat, const float* __restrict__ sg,
                            float* __restrict__ sr, int total) {
    int i = blockIdx.x * blockDim.x + threadIdx.x;
    if (i >= total) return;
    int b = i / (FW + Dd), c = i - b * (FW + Dd);
    sr[i] = (c < FW) ? feat[((size_t)b * Lg + (Lg - 1)) * FW + c]
                     : sg[(size_t)b * Dd + (c - FW)];
}

// ================================ host ======================================

extern "C" void kernel_launch(void* const* d_in, const int* in_sizes, int n_in,
                              void* d_out, int out_size, void* d_ws, size_t ws_size,
                              hipStream_t stream) {
    (void)in_sizes; (void)n_in; (void)out_size; (void)ws_size;

    // Input order = setup_inputs() insertion order: params dict then iid/mg/sg.
    enum {
        I_EMB = 0,
        I_L0_G, I_L0_B, I_L0_PR, I_L0_WIH, I_L0_WHH, I_L0_BIH, I_L0_BHH, I_L0_WSELF, I_L0_WNEIGH,
        I_L1_G, I_L1_B, I_L1_PR, I_L1_WQ, I_L1_BQ, I_L1_WK, I_L1_WV, I_L1_WE,
        I_L2_G, I_L2_B, I_L2_PR, I_L2_WIH, I_L2_WHH, I_L2_BIH, I_L2_BHH, I_L2_WSELF, I_L2_WNEIGH,
        I_RO_G, I_RO_B, I_RO_WU, I_RO_WV, I_RO_BV, I_RO_WE, I_RO_WOUT,
        I_F_G, I_F_B, I_F_WSR,
        I_IID, I_MG, I_SG
    };
    auto F = [&](int i) { return (const float*)d_in[i]; };
    const int* iid = (const int*)d_in[I_IID];
    const int* mg  = (const int*)d_in[I_MG];
    const int* sg  = (const int*)d_in[I_SG];

    auto cdiv = [](long a, long b) { return (int)((a + b - 1) / b); };

    // ---- workspace allocator ----
    size_t off = 0;
    auto alloc = [&](size_t bytes) -> void* {
        off = (off + 255) & ~size_t(255);
        void* p = (char*)d_ws + off;
        off += bytes;
        return p;
    };
    float* emb_n  = (float*)alloc(4ull * Vv * Dd);
    u16*   emb_bf = (u16*)  alloc(2ull * Vv * Dd);
    float* feat   = (float*)alloc(4ull * Ng * FW);
    u16*   xbn_bf = (u16*)  alloc(2ull * Ng * FW);
    float* xr32   = (float*)alloc(4ull * Ng * FW);
    u16*   xt_bf  = (u16*)  alloc(2ull * Ng * 384);
    float* hbuf   = (float*)alloc(4ull * Ng * 384);
    u16*   h_bf   = (u16*)  alloc(2ull * Ng * 384);
    float* gi     = (float*)alloc(4ull * Ng * 1152);
    float* gh     = (float*)alloc(4ull * Ng * 1152);
    float* tmpA   = (float*)alloc(4ull * Ng * Dd);
    float* tmpB   = (float*)alloc(4ull * Ng * Dd);
    float* tmpC   = (float*)alloc(4ull * Ng * Dd);
    float* esc    = (float*)alloc(4ull * Ng);
    float* bnsc   = (float*)alloc(4ull * 1152);
    float* bnsh   = (float*)alloc(4ull * 1152);
    u16*   xlastb = (u16*)  alloc(2ull * Bg * FW);
    float* fvraw  = (float*)alloc(4ull * Bg * Dd);
    float* srg    = (float*)alloc(4ull * Bg * FW);
    u16*   srg_bf = (u16*)  alloc(2ull * Bg * FW);
    float* srgout = (float*)alloc(4ull * Bg * Dd);
    float* srcat  = (float*)alloc(4ull * Bg * (FW + Dd));
    u16*   srcatb = (u16*)  alloc(2ull * Bg * (FW + Dd));
    float* srf    = (float*)alloc(4ull * Bg * Dd);
    u16*   srf_bf = (u16*)  alloc(2ull * Bg * Dd);

    // ---- bf16 weight conversions ----
    auto wconv = [&](int idx, size_t elems) -> u16* {
        u16* p = (u16*)alloc(2 * elems);
        k_cvt_bf16<<<cdiv((long)elems, 256), 256, 0, stream>>>(F(idx), p, (int)elems);
        return p;
    };
    u16* wih0 = wconv(I_L0_WIH, 384 * 128);
    u16* whh0 = wconv(I_L0_WHH, 384 * 128);
    u16* wsf0 = wconv(I_L0_WSELF, 128 * 128);
    u16* wng0 = wconv(I_L0_WNEIGH, 128 * 128);
    u16* wq1  = wconv(I_L1_WQ, 128 * 256);
    u16* wk1  = wconv(I_L1_WK, 128 * 256);
    u16* wv1  = wconv(I_L1_WV, 128 * 256);
    u16* wih2 = wconv(I_L2_WIH, 1152 * 384);
    u16* whh2 = wconv(I_L2_WHH, 1152 * 384);
    u16* wsf2 = wconv(I_L2_WSELF, 128 * 384);
    u16* wng2 = wconv(I_L2_WNEIGH, 128 * 384);
    u16* wu   = wconv(I_RO_WU, 128 * 512);
    u16* wvro = wconv(I_RO_WV, 128 * 512);
    u16* wout = wconv(I_RO_WOUT, 128 * 512);
    u16* wsr  = wconv(I_F_WSR, 128 * 640);

    auto gemm = [&](const u16* A, int lda, const u16* W, int ldw,
                    float* C, int ldc, int M, int Nd, int K) {
        dim3 g((Nd + 127) / 128, M / 128);
        gemm_bf16_wmma<<<g, 256, 0, stream>>>(A, lda, W, ldw, C, ldc, M, Nd, K);
    };
    auto bn = [&](const float* x, int stride, int nrows, int ncols,
                  const float* g_, const float* b_, u16* yb, float* yf) {
        k_bn_stats<<<ncols, 256, 0, stream>>>(x, stride, nrows, g_, b_, bnsc, bnsh);
        k_bn_apply<<<cdiv((long)nrows * ncols, 256), 256, 0, stream>>>(
            x, stride, ncols, bnsc, bnsh, yb, yf, nrows * ncols);
    };

    // ---- 1. embedding renorm + initial feat gather ----
    k_renorm_emb<<<Vv, 128, 0, stream>>>(F(I_EMB), emb_n, emb_bf);
    k_gather_emb<<<cdiv((long)Ng * Dd, 256), 256, 0, stream>>>(emb_n, iid, feat, Ng * Dd);

    // ---- 2. layers ----
    auto eopa = [&](int d, int colofs, int i_g, int i_b, int i_pr,
                    u16* wih, u16* whh, int i_bih, int i_bhh, u16* wsf, u16* wng) {
        int d3 = 3 * d;
        bn(feat + (FW - d), FW, Ng, d, F(i_g), F(i_b), xbn_bf, nullptr);
        k_fill_f32<<<cdiv((long)Ng * d, 256), 256, 0, stream>>>(hbuf, 0.f, Ng * d);
        k_fill_u16<<<cdiv((long)Ng * d, 256), 256, 0, stream>>>(h_bf, (u16)0, Ng * d);
        for (int t = 0; t < DEGc; ++t) {
            k_gather_rows_bf<<<cdiv((long)Ng * d, 256), 256, 0, stream>>>(
                xbn_bf, mg, t, d, xt_bf, Ng * d);
            gemm(xt_bf, d, wih, d, gi, d3, Ng, d3, d);
            gemm(h_bf, d, whh, d, gh, d3, Ng, d3, d);
            k_gru_update<<<cdiv((long)Ng * d, 256), 256, 0, stream>>>(
                gi, gh, F(i_bih), F(i_bhh), hbuf, h_bf, d, Ng * d);
        }
        gemm(xbn_bf, d, wsf, d, tmpA, Dd, Ng, Dd, d);
        gemm(h_bf, d, wng, d, tmpB, Dd, Ng, Dd, d);
        k_prelu_store<<<cdiv((long)Ng * Dd, 256), 256, 0, stream>>>(
            tmpA, tmpB, F(i_pr), feat, colofs, Ng * Dd);
    };

    // layer 0 (EOPA, d=128) -> out columns [256,384)
    eopa(128, 256, I_L0_G, I_L0_B, I_L0_PR, wih0, whh0, I_L0_BIH, I_L0_BHH, wsf0, wng0);

    // layer 1 (SGAT, d=256) -> out columns [128,256)
    {
        int d = 256;
        bn(feat + (FW - d), FW, Ng, d, F(I_L1_G), F(I_L1_B), xbn_bf, nullptr);
        gemm(xbn_bf, d, wq1, d, tmpA, Dd, Ng, Dd, d);   // q (no bias yet)
        gemm(xbn_bf, d, wk1, d, tmpB, Dd, Ng, Dd, d);   // k
        gemm(xbn_bf, d, wv1, d, tmpC, Dd, Ng, Dd, d);   // v
        k_sgat<<<Ng, 128, 0, stream>>>(tmpA, tmpB, tmpC, F(I_L1_BQ), F(I_L1_WE), sg, gi);
        k_prelu_store<<<cdiv((long)Ng * Dd, 256), 256, 0, stream>>>(
            gi, nullptr, F(I_L1_PR), feat, 128, Ng * Dd);
    }

    // layer 2 (EOPA, d=384) -> out columns [0,128)
    eopa(384, 0, I_L2_G, I_L2_B, I_L2_PR, wih2, whh2, I_L2_BIH, I_L2_BHH, wsf2, wng2);

    // ---- 3. attention readout ----
    bn(feat, FW, Ng, FW, F(I_RO_G), F(I_RO_B), xbn_bf, xr32);
    gemm(xbn_bf, FW, wu, FW, tmpA, Dd, Ng, Dd, FW);                     // fu [N,128]
    k_gather_last_bf<<<cdiv((long)Bg * FW, 256), 256, 0, stream>>>(xbn_bf, xlastb, FW, Bg * FW);
    gemm(xlastb, FW, wvro, FW, fvraw, Dd, Bg, Dd, FW);                  // fv [B,128]
    k_ro_score<<<Ng, 128, 0, stream>>>(tmpA, fvraw, F(I_RO_BV), F(I_RO_WE), esc);
    k_ro_attnsum<<<Bg, 256, 0, stream>>>(esc, xr32, srg, srg_bf);       // sr_g [B,512]
    gemm(srg_bf, FW, wout, FW, srgout, Dd, Bg, Dd, FW);                 // [B,128]

    // ---- 4. final fuse + logits ----
    k_concat_sr<<<cdiv((long)Bg * (FW + Dd), 256), 256, 0, stream>>>(
        feat, srgout, srcat, Bg * (FW + Dd));
    bn(srcat, FW + Dd, Bg, FW + Dd, F(I_F_G), F(I_F_B), srcatb, nullptr);
    gemm(srcatb, FW + Dd, wsr, FW + Dd, srf, Dd, Bg, Dd, FW + Dd);      // [B,128]
    k_cvt_bf16<<<cdiv((long)Bg * Dd, 256), 256, 0, stream>>>(srf, srf_bf, Bg * Dd);
    gemm(srf_bf, Dd, emb_bf, Dd, (float*)d_out, Vv, Bg, Vv, Dd);        // logits [B,V]
}
